// Predict_model_65704409694775
// MI455X (gfx1250) — matrix operand order repaired and checked by reading, not compiled, not run on hardware
//
#include <hip/hip_runtime.h>
#include <hip/hip_bf16.h>
#include <math.h>
#include <stdint.h>

// Problem constants from the reference
#define Tg   5
#define Nn   20000
#define Ee   320000
#define Dd   256
#define Hh   8
#define NHIDc 32
#define BNb  4096
#define BCc  2048
#define G4   1024   // 4*D LSTM gate width

typedef __attribute__((ext_vector_type(16))) __bf16 v16bf;
typedef __attribute__((ext_vector_type(8)))  float  v8f;

static __device__ __forceinline__ float sigmf(float x) { return 1.f / (1.f + expf(-x)); }

__device__ __forceinline__ float atomicMaxFloat(float* addr, float val) {
  int* ia = (int*)addr;
  int old = __float_as_int(*addr);
  while (__int_as_float(old) < val) {
    int assumed = old;
    old = atomicCAS(ia, assumed, __float_as_int(val));
    if (old == assumed) break;
  }
  return __int_as_float(old);
}

// ---------------------------------------------------------------------------
// WMMA bf16 GEMM: Out[M x N] = A[M x K](bf16) * Bt^T (+Cin) (+bias)
//   A  : M x K row-major bf16
//   Bt : N x K row-major bf16 (B transposed -> contiguous K per output column)
// Block tile 64x128, BK=32, 8 waves; K must be a multiple of 32 (K=256 here).
// Global->LDS staging uses async copies (GLOBAL_LOAD_ASYNC_TO_LDS_B128) with
// double buffering: tile k+1 streams in while WMMA consumes tile k.
// ---------------------------------------------------------------------------
#define BM 64
#define BN 128
#define BK 32
#define LDS_ROW (BK + 8)                       // 40 elems, 80B row stride
#define ASZ (BM * LDS_ROW * 2)                 // 5120 B
#define BSZ (BN * LDS_ROW * 2)                 // 10240 B
#define BUFSZ (ASZ + BSZ)                      // 15360 B  (x2 buffers = 30720)

__device__ __forceinline__ void async_b128(uint32_t lds_off, const void* gaddr) {
  asm volatile("global_load_async_to_lds_b128 %0, %1, off"
               :: "v"(lds_off), "v"(gaddr) : "memory");
}

__global__ __launch_bounds__(256) void gemm_bf16_wmma(
    const __bf16* __restrict__ A, const __bf16* __restrict__ Bt,
    const float* __restrict__ Cin, const float* __restrict__ bias,
    float* __restrict__ Out, int M, int N, int K,
    int lda, int ldbt, int ldcin, int ldc) {
  extern __shared__ char smem[];  // 2 * BUFSZ bytes, dynamic => LDS offset 0
  const int tid  = threadIdx.x;
  const int lane = tid & 31;
  const int wave = tid >> 5;
  const int m0 = blockIdx.x * BM;
  const int n0 = blockIdx.y * BN;
  const int wm = (wave & 3) * 16;
  const int wn = (wave >> 2) * 64;

  const int crow = (lane >> 4) * 8;  // C/D: M = r + 8*(lane>=16)
  const int ccol = lane & 15;

  // per-thread copy coordinates (branchless: clamp row, tail rows are
  // duplicates whose products land only in rows the store masks off)
  const int ar = tid >> 2, ac = (tid & 3) * 8;   // A: 16B per thread
  const int bn = tid >> 1, bc2 = (tid & 1) * 16; // B: 32B per thread
  int agr = m0 + ar; if (agr > M - 1) agr = M - 1;
  const __bf16* gA = A + (long)agr * lda + ac;
  const __bf16* gB = Bt + (long)(n0 + bn) * ldbt + bc2;
  const uint32_t laA = (uint32_t)((ar * LDS_ROW + ac) * 2);
  const uint32_t laB = (uint32_t)(ASZ + (bn * LDS_ROW + bc2) * 2);

  v8f acc[4];
#pragma unroll
  for (int f = 0; f < 4; ++f) {
    if (Cin) {
      int gr = m0 + wm + crow;
      if (gr > M - 1) gr = M - 1;  // clamp; masked at store
#pragma unroll
      for (int r = 0; r < 8; ++r) {
        int rr = gr + r; if (rr > M - 1) rr = M - 1;
        acc[f][r] = Cin[(long)rr * ldcin + n0 + wn + f * 16 + ccol];
      }
    } else {
#pragma unroll
      for (int r = 0; r < 8; ++r) acc[f][r] = 0.f;
    }
  }

  // prologue: stream tile 0 into buffer 0
  async_b128(laA, gA);
  async_b128(laB, gB);
  async_b128(laB + 16, (const char*)gB + 16);

  const int nk = K / BK;
  for (int kt = 0; kt < nk; ++kt) {
    const int buf = kt & 1;
    if (kt + 1 < nk) {  // stream tile kt+1 into the other buffer
      const uint32_t o = (uint32_t)((buf ^ 1) * BUFSZ);
      const long koff = (long)(kt + 1) * BK;
      async_b128(o + laA, gA + koff);
      async_b128(o + laB, gB + koff);
      async_b128(o + laB + 16, (const char*)(gB + koff) + 16);
      asm volatile("s_wait_asynccnt 0x3" ::: "memory");  // tile kt landed
    } else {
      asm volatile("s_wait_asynccnt 0x0" ::: "memory");
    }
    __syncthreads();

    const __bf16* Asb = (const __bf16*)(smem + buf * BUFSZ);
    const __bf16* Bsb = (const __bf16*)(smem + buf * BUFSZ + ASZ);

    // A fragment (16x32 bf16, ISA layout): 2x contiguous 16B runs
    v16bf av;
    {
      int arow = wm + (lane & 15);
      int kb = (lane >> 4) * 8;
      const __bf16* p = Asb + arow * LDS_ROW;
#pragma unroll
      for (int j = 0; j < 8; ++j) {
        av[j]     = p[kb + j];
        av[j + 8] = p[kb + 16 + j];
      }
    }
#pragma unroll
    for (int f = 0; f < 4; ++f) {
      // B fragment: lane holds col n = lane&15, K = (lane>>4)*8 + j and 16 + (lane>>4)*8 + j
      v16bf bv;
      const __bf16* p = Bsb + (wn + f * 16 + (lane & 15)) * LDS_ROW + (lane >> 4) * 8;
#pragma unroll
      for (int j = 0; j < 8; ++j) { bv[j] = p[j]; bv[j + 8] = p[16 + j]; }
      acc[f] = __builtin_amdgcn_wmma_f32_16x16x32_bf16(
          false, av, false, bv, (short)0, acc[f], false, false);
    }
    __syncthreads();
  }

#pragma unroll
  for (int f = 0; f < 4; ++f) {
#pragma unroll
    for (int r = 0; r < 8; ++r) {
      int gr = m0 + wm + crow + r;
      int gc = n0 + wn + f * 16 + ccol;
      if (gr < M) {
        float v = acc[f][r];
        if (bias) v += bias[gc];
        Out[(long)gr * ldc + gc] = v;
      }
    }
  }
}

// ---------------------------------------------------------------------------
// Elementwise / reduction kernels
// ---------------------------------------------------------------------------
__global__ void fill_f32(float* p, float v, int n) {
  int i = blockIdx.x * blockDim.x + threadIdx.x;
  if (i < n) p[i] = v;
}
__global__ void fill_i32(int* p, int v, int n) {
  int i = blockIdx.x * blockDim.x + threadIdx.x;
  if (i < n) p[i] = v;
}
__global__ void f32_to_bf16_k(const float* __restrict__ x, __bf16* __restrict__ y, int n) {
  int i = blockIdx.x * blockDim.x + threadIdx.x;
  if (i < n) y[i] = (__bf16)x[i];
}
// W[R x C] row-major -> WT[C x R] bf16
__global__ void transpose_to_bf16(const float* __restrict__ W, __bf16* __restrict__ WT,
                                  int R, int C) {
  int i = blockIdx.x * blockDim.x + threadIdx.x;
  if (i < R * C) {
    int r = i / C, c = i % C;
    WT[(long)c * R + r] = (__bf16)W[i];
  }
}
__global__ void bias_sum_k(const float* a, const float* b, float* o, int n) {
  int i = blockIdx.x * blockDim.x + threadIdx.x;
  if (i < n) o[i] = a[i] + b[i];
}

// per-column mean / rsqrt(var+eps) over nrows (stride Dd)
__global__ void colstats_k(const float* __restrict__ x, int nrows, float* mean, float* rstd) {
  int col = blockIdx.x;
  float s = 0.f, s2 = 0.f;
  for (int r = threadIdx.x; r < nrows; r += blockDim.x) {
    float v = x[(long)r * Dd + col];
    s += v; s2 += v * v;
  }
  __shared__ float sh[256], sh2[256];
  sh[threadIdx.x] = s; sh2[threadIdx.x] = s2;
  __syncthreads();
  for (int off = 128; off > 0; off >>= 1) {
    if (threadIdx.x < off) { sh[threadIdx.x] += sh[threadIdx.x + off]; sh2[threadIdx.x] += sh2[threadIdx.x + off]; }
    __syncthreads();
  }
  if (threadIdx.x == 0) {
    float mu = sh[0] / (float)nrows;
    float var = sh2[0] / (float)nrows - mu * mu;
    mean[col] = mu;
    rstd[col] = rsqrtf(var + 1e-5f);
  }
}
__global__ void bn_apply_bf16_k(const float* __restrict__ x, const float* mean, const float* rstd,
                                const float* g, const float* b, __bf16* __restrict__ y, int n) {
  int i = blockIdx.x * blockDim.x + threadIdx.x;
  if (i < n) {
    int c = i & (Dd - 1);
    y[i] = (__bf16)((x[i] - mean[c]) * rstd[c] * g[c] + b[c]);
  }
}

// attention projections: ps/pd[n,h] = dot(h[n,h,:], a_src/dst[h,:])
__global__ void node_proj_k(const float* __restrict__ hf, const float* __restrict__ asrc,
                            const float* __restrict__ adst, float* ps, float* pd, int n) {
  int i = blockIdx.x * blockDim.x + threadIdx.x;
  if (i >= n * Hh) return;
  int node = i >> 3, hh = i & 7;
  const float* hp = hf + (long)node * Dd + hh * NHIDc;
  float s = 0.f, d = 0.f;
#pragma unroll
  for (int k = 0; k < NHIDc; ++k) { s += hp[k] * asrc[hh * NHIDc + k]; d += hp[k] * adst[hh * NHIDc + k]; }
  ps[i] = s; pd[i] = d;
}
__global__ void edge_max_k(const int* __restrict__ src, const int* __restrict__ dst,
                           const float* ps, const float* pd, float* logits, float* m) {
  int i = blockIdx.x * blockDim.x + threadIdx.x;
  if (i >= Ee * Hh) return;
  int e = i >> 3, hh = i & 7;
  float lg = ps[src[e] * Hh + hh] + pd[dst[e] * Hh + hh];
  lg = (lg > 0.f) ? lg : 0.2f * lg;  // leaky_relu(0.2)
  logits[i] = lg;
  atomicMaxFloat(&m[dst[e] * Hh + hh], lg);
}
__global__ void edge_exp_k(const int* __restrict__ dst, float* alpha, const float* m, float* ssum) {
  int i = blockIdx.x * blockDim.x + threadIdx.x;
  if (i >= Ee * Hh) return;
  int e = i >> 3, hh = i & 7;
  float ex = expf(alpha[i] - m[dst[e] * Hh + hh]);
  alpha[i] = ex;
  atomicAdd(&ssum[dst[e] * Hh + hh], ex);
}
__global__ void alpha_norm_k(const int* __restrict__ dst, float* alpha, const float* ssum) {
  int i = blockIdx.x * blockDim.x + threadIdx.x;
  if (i >= Ee * Hh) return;
  int e = i >> 3, hh = i & 7;
  alpha[i] = alpha[i] / (ssum[dst[e] * Hh + hh] + 1e-16f);
}
// per-edge aggregation: one 32-lane group per edge, 8 floats per lane
__global__ void aggregate_k(const int* __restrict__ src, const int* __restrict__ dst,
                            const float* __restrict__ alpha, const float* __restrict__ hf,
                            float* __restrict__ out) {
  long gid = (long)blockIdx.x * blockDim.x + threadIdx.x;
  if (gid >= (long)Ee * 32) return;
  int e = (int)(gid >> 5), l = (int)(gid & 31);
  int s = src[e], t = dst[e];
#pragma unroll
  for (int hh = 0; hh < Hh; ++hh) {
    int d = hh * NHIDc + l;
    atomicAdd(&out[(long)t * Dd + d], alpha[e * Hh + hh] * hf[(long)s * Dd + d]);
  }
}
__global__ void store_embeds_k(const float* __restrict__ src, float* __restrict__ embeds, int tslot) {
  int i = blockIdx.x * blockDim.x + threadIdx.x;
  if (i >= Nn * Dd) return;
  int node = i >> 8, j = i & 255;
  embeds[((long)node * Tg + tslot) * Dd + j] = src[i];
}
__global__ void gather_temp_k(const float* __restrict__ embeds, const int* __restrict__ nb,
                              float* __restrict__ temp) {
  int i = blockIdx.x * blockDim.x + threadIdx.x;
  if (i >= BNb * Tg * Dd) return;
  int b = i / (Tg * Dd);
  int rem = i - b * Tg * Dd;
  temp[i] = embeds[(long)nb[b] * Tg * Dd + rem];
}
__global__ void flags_k(const float* __restrict__ temp, int* __restrict__ flags) {
  int i = blockIdx.x * blockDim.x + threadIdx.x;
  if (i >= BNb * Tg) return;
  const float* p = temp + (long)i * Dd;
  int f = 0;
  for (int j = 0; j < Dd; ++j)
    if (p[j] != 0.f) { f = 1; break; }
  flags[i] = f;
}
__global__ void lens_k(const int* __restrict__ flags, int* __restrict__ lens, int* maxlen) {
  int b = blockIdx.x * blockDim.x + threadIdx.x;
  if (b >= BNb) return;
  int s = 0;
#pragma unroll
  for (int t = 0; t < Tg; ++t) s += flags[b * Tg + t];
  if (s > Tg) s = Tg;
  lens[b] = s;
  atomicMax(maxlen, s);
}
__global__ void rev_within_k(const float* __restrict__ x, const int* __restrict__ lens,
                             float* __restrict__ o) {
  int i = blockIdx.x * blockDim.x + threadIdx.x;
  if (i >= BNb * Tg * Dd) return;
  int b = i / (Tg * Dd);
  int rem = i - b * Tg * Dd;
  int t = rem >> 8, j = rem & 255;
  int it = lens[b] - 1 - t;
  it = it < 0 ? 0 : (it > Tg - 1 ? Tg - 1 : it);
  o[i] = x[((long)b * Tg + it) * Dd + j];
}
__global__ void lstm_gate_k(const float* __restrict__ gbuf, float* __restrict__ c,
                            float* __restrict__ hs, __bf16* __restrict__ hbf, int t) {
  int i = blockIdx.x * blockDim.x + threadIdx.x;
  if (i >= BNb * Dd) return;
  int b = i >> 8, j = i & 255;
  long base = (long)b * G4;
  float gi = sigmf(gbuf[base + j]);
  float gf = sigmf(gbuf[base + 256 + j]);
  float gg = tanhf(gbuf[base + 512 + j]);
  float go = sigmf(gbuf[base + 768 + j]);
  float cn = gf * c[i] + gi * gg;
  c[i] = cn;
  float h = go * tanhf(cn);
  hs[((long)b * Tg + t) * Dd + j] = h;
  hbf[i] = (__bf16)h;
}
__global__ void pooled_k(const float* __restrict__ hsF, const float* __restrict__ hsB,
                         const int* __restrict__ lens, const int* __restrict__ maxlen,
                         float* __restrict__ pooled) {
  int i = blockIdx.x * blockDim.x + threadIdx.x;
  if (i >= BNb * Dd) return;
  int b = i >> 8, j = i & 255;
  int L = lens[b];
  int ml = *maxlen; if (ml < 1) ml = 1;
  float denom = (float)ml;
  float sum = 0.f;
#pragma unroll
  for (int t = 0; t < Tg; ++t) {
    if (t < L) {
      int it = L - 1 - t;
      it = it < 0 ? 0 : (it > Tg - 1 ? Tg - 1 : it);
      float fb = (it < L) ? hsB[((long)b * Tg + it) * Dd + j] : 0.f;
      sum += 0.5f * (hsF[((long)b * Tg + t) * Dd + j] + fb);
    }
  }
  pooled[i] = sum / denom;
}
__global__ void scatter_k(const float* __restrict__ pooled, const int* __restrict__ nb,
                          float* __restrict__ emb) {
  int i = blockIdx.x * blockDim.x + threadIdx.x;
  if (i >= BNb * Dd) return;
  int b = i >> 8, j = i & 255;
  emb[(long)nb[b] * Dd + j] = pooled[i];
}
__global__ void atts_k(const int* __restrict__ dst, const float* __restrict__ att,
                       float* __restrict__ atts, float* __restrict__ maxv) {
  int e = blockIdx.x * blockDim.x + threadIdx.x;
  if (e >= Ee) return;
  float s = 0.f;
#pragma unroll
  for (int hh = 0; hh < Hh; ++hh) s += att[e * Hh + hh];
  atts[e] = s;
  atomicMaxFloat(&maxv[dst[e]], s);
}
__global__ void argmin_k(const int* __restrict__ dst, const float* __restrict__ atts,
                         const float* __restrict__ maxv, int* __restrict__ argm) {
  int e = blockIdx.x * blockDim.x + threadIdx.x;
  if (e >= Ee) return;
  int t = dst[e];
  if (atts[e] == maxv[t]) atomicMin(&argm[t], e);
}
__global__ void srcsel_k(const int* __restrict__ argm, const int* __restrict__ nodes,
                         const int* __restrict__ srcp, float* __restrict__ out) {
  int i = blockIdx.x * blockDim.x + threadIdx.x;
  if (i >= BCc) return;
  int a = argm[nodes[i]];
  if (a < 0) a = 0;
  if (a > Ee - 1) a = Ee - 1;
  out[i] = (float)srcp[a];
}
__global__ void gather_rows_k(const float* __restrict__ x, const int* __restrict__ nodes,
                              float* __restrict__ o) {
  int i = blockIdx.x * blockDim.x + threadIdx.x;
  if (i >= BCc * Dd) return;
  int b = i >> 8, j = i & 255;
  o[i] = x[(long)nodes[b] * Dd + j];
}
__global__ void mlp_k(const float* __restrict__ hclf, const float* mean, const float* rstd,
                      const float* g2, const float* b2n,
                      const float* W1, const float* b1, const float* W2, const float* b2,
                      const float* W3, const float* b3, float* __restrict__ out) {
  int bId = blockIdx.x;
  int tid = threadIdx.x;
  __shared__ float s0[256];
  __shared__ float s1[32];
  __shared__ float s2[16];
  float v = hclf[(long)bId * Dd + tid];
  s0[tid] = (v - mean[tid]) * rstd[tid] * g2[tid] + b2n[tid];
  __syncthreads();
  if (tid < 32) {
    float a = b1[tid];
    for (int k = 0; k < 256; ++k) a += s0[k] * W1[k * 32 + tid];
    s1[tid] = tanhf(a);
  }
  __syncthreads();
  if (tid < 16) {
    float a = b2[tid];
    for (int k = 0; k < 32; ++k) a += s1[k] * W2[k * 16 + tid];
    s2[tid] = tanhf(a);
  }
  __syncthreads();
  if (tid == 0) {
    float a = b3[0];
    for (int k = 0; k < 16; ++k) a += s2[k] * W3[k];
    out[bId] = 1.f / (1.f + expf(-a));
  }
}

// ---------------------------------------------------------------------------
static inline int cdiv(long a, long b) { return (int)((a + b - 1) / b); }

extern "C" void kernel_launch(void* const* d_in, const int* in_sizes, int n_in,
                              void* d_out, int out_size, void* d_ws, size_t ws_size,
                              hipStream_t stream) {
  const float* emb      = (const float*)d_in[0];
  const int*   edges    = (const int*)d_in[1];
  const int*   clf_nb   = (const int*)d_in[4];
  const int*   clf_nd   = (const int*)d_in[5];
  const float* bn1_g    = (const float*)d_in[6];
  const float* bn1_b    = (const float*)d_in[7];
  const float* gnn_W    = (const float*)d_in[8];
  const float* gnn_b    = (const float*)d_in[9];
  const float* gnn_asrc = (const float*)d_in[10];
  const float* gnn_adst = (const float*)d_in[11];
  const float* lstm_Wih = (const float*)d_in[12];
  const float* lstm_Whh = (const float*)d_in[13];
  const float* lstm_bih = (const float*)d_in[14];
  const float* lstm_bhh = (const float*)d_in[15];
  const float* bn2_g    = (const float*)d_in[16];
  const float* bn2_b    = (const float*)d_in[17];
  const float* clf_W    = (const float*)d_in[18];
  const float* clf_b    = (const float*)d_in[19];
  const float* clf_asrc = (const float*)d_in[20];
  const float* clf_adst = (const float*)d_in[21];
  const float* W1 = (const float*)d_in[22];
  const float* b1 = (const float*)d_in[23];
  const float* W2 = (const float*)d_in[24];
  const float* b2 = (const float*)d_in[25];
  const float* W3 = (const float*)d_in[26];
  const float* b3 = (const float*)d_in[27];
  float* outp = (float*)d_out;
  (void)in_sizes; (void)n_in; (void)out_size; (void)ws_size;

  char* wp = (char*)d_ws;
  auto alloc = [&](size_t bytes) -> void* {
    void* r = (void*)wp;
    wp += (bytes + 255) & ~(size_t)255;
    return r;
  };

  float*  bufA   = (float*)alloc((size_t)Nn * Dd * 4);
  float*  bufB   = (float*)alloc((size_t)Nn * Dd * 4);
  float*  hfeat  = (float*)alloc((size_t)Nn * Dd * 4);
  __bf16* xnbf   = (__bf16*)alloc((size_t)Nn * Dd * 2);
  __bf16* wbf0   = (__bf16*)alloc((size_t)Dd * Dd * 2);   // gnn_W[0]^T  (N x K)
  __bf16* wbf1   = (__bf16*)alloc((size_t)Dd * Dd * 2);   // gnn_W[1]^T
  __bf16* wbfC   = (__bf16*)alloc((size_t)Dd * Dd * 2);   // clf_W^T
  __bf16* wihB   = (__bf16*)alloc((size_t)2 * G4 * Dd * 2); // Wih as-is = (N x K)
  __bf16* whhB   = (__bf16*)alloc((size_t)2 * G4 * Dd * 2); // Whh as-is = (N x K)
  float*  bsum   = (float*)alloc((size_t)2 * G4 * 4);
  float*  ps     = (float*)alloc((size_t)Nn * Hh * 4);
  float*  pd     = (float*)alloc((size_t)Nn * Hh * 4);
  float*  mM     = (float*)alloc((size_t)Nn * Hh * 4);
  float*  sS     = (float*)alloc((size_t)Nn * Hh * 4);
  float*  alpha  = (float*)alloc((size_t)Ee * Hh * 4);
  float*  embeds = (float*)alloc((size_t)Nn * Tg * Dd * 4);
  float*  temp   = (float*)alloc((size_t)BNb * Tg * Dd * 4);
  float*  xrev   = (float*)alloc((size_t)BNb * Tg * Dd * 4);
  __bf16* xbf    = (__bf16*)alloc((size_t)BNb * Tg * Dd * 2);
  int*    flags  = (int*)alloc((size_t)BNb * Tg * 4);
  int*    lens   = (int*)alloc((size_t)BNb * 4);
  int*    maxlen = (int*)alloc(256);
  float*  gbuf   = (float*)alloc((size_t)BNb * G4 * 4);
  float*  cbuf   = (float*)alloc((size_t)BNb * Dd * 4);
  __bf16* hbf    = (__bf16*)alloc((size_t)BNb * Dd * 2);
  float*  hsF    = (float*)alloc((size_t)BNb * Tg * Dd * 4);
  float*  hsB    = (float*)alloc((size_t)BNb * Tg * Dd * 4);
  float*  pooled = (float*)alloc((size_t)BNb * Dd * 4);
  float*  meanb  = (float*)alloc(256 * 4);
  float*  rstdb  = (float*)alloc(256 * 4);
  float*  atts   = (float*)alloc((size_t)Ee * 4);
  float*  maxv   = (float*)alloc((size_t)Nn * 4);
  int*    argm   = (int*)alloc((size_t)Nn * 4);
  float*  hclf   = (float*)alloc((size_t)BCc * Dd * 4);

  const int TB = 256;
  const size_t GEMM_LDS = 2 * BUFSZ;  // double-buffered tiles

  // ---- weight conversions (once per call) ----
  transpose_to_bf16<<<cdiv(Dd * Dd, TB), TB, 0, stream>>>(gnn_W, wbf0, Dd, Dd);
  transpose_to_bf16<<<cdiv(Dd * Dd, TB), TB, 0, stream>>>(gnn_W + Dd * Dd, wbf1, Dd, Dd);
  transpose_to_bf16<<<cdiv(Dd * Dd, TB), TB, 0, stream>>>(clf_W, wbfC, Dd, Dd);
  f32_to_bf16_k<<<cdiv(2L * G4 * Dd, TB), TB, 0, stream>>>(lstm_Wih, wihB, 2 * G4 * Dd);
  f32_to_bf16_k<<<cdiv(2L * G4 * Dd, TB), TB, 0, stream>>>(lstm_Whh, whhB, 2 * G4 * Dd);
  bias_sum_k<<<cdiv(2 * G4, TB), TB, 0, stream>>>(lstm_bih, lstm_bhh, bsum, 2 * G4);

  dim3 gemmN(cdiv(Nn, BM), Dd / BN);     // 313 x 2
  dim3 gemmL(cdiv(BNb, BM), G4 / BN);    // 64 x 8

  // ---- GNN: T graphs x L layers ----
  for (int g = 0; g < Tg; ++g) {
    const float* ein = emb + (size_t)g * Nn * Dd;
    const int* srcp = edges + (size_t)g * 2 * Ee;
    const int* dstp = srcp + Ee;
    for (int l = 0; l < 2; ++l) {
      colstats_k<<<Dd, TB, 0, stream>>>(ein, Nn, meanb, rstdb);
      bn_apply_bf16_k<<<cdiv((long)Nn * Dd, TB), TB, 0, stream>>>(ein, meanb, rstdb, bn1_g, bn1_b, xnbf, Nn * Dd);
      gemm_bf16_wmma<<<gemmN, TB, GEMM_LDS, stream>>>(xnbf, (l == 0) ? wbf0 : wbf1, nullptr,
                                                      gnn_b + (size_t)l * Dd, hfeat,
                                                      Nn, Dd, Dd, Dd, Dd, 0, Dd);
      node_proj_k<<<cdiv((long)Nn * Hh, TB), TB, 0, stream>>>(hfeat, gnn_asrc + (size_t)l * Hh * NHIDc,
                                                              gnn_adst + (size_t)l * Hh * NHIDc, ps, pd, Nn);
      fill_f32<<<cdiv((long)Nn * Hh, TB), TB, 0, stream>>>(mM, -1e30f, Nn * Hh);
      hipMemsetAsync(sS, 0, (size_t)Nn * Hh * 4, stream);
      edge_max_k<<<cdiv((long)Ee * Hh, TB), TB, 0, stream>>>(srcp, dstp, ps, pd, alpha, mM);
      edge_exp_k<<<cdiv((long)Ee * Hh, TB), TB, 0, stream>>>(dstp, alpha, mM, sS);
      alpha_norm_k<<<cdiv((long)Ee * Hh, TB), TB, 0, stream>>>(dstp, alpha, sS);
      float* outb = (l == 0) ? bufA : bufB;
      hipMemsetAsync(outb, 0, (size_t)Nn * Dd * 4, stream);
      aggregate_k<<<cdiv((long)Ee * 32, TB), TB, 0, stream>>>(srcp, dstp, alpha, hfeat, outb);
      ein = outb;
    }
    store_embeds_k<<<cdiv((long)Nn * Dd, TB), TB, 0, stream>>>(ein, embeds, Tg - 1 - g);
  }

  // ---- LSTM prep ----
  gather_temp_k<<<cdiv((long)BNb * Tg * Dd, TB), TB, 0, stream>>>(embeds, clf_nb, temp);
  flags_k<<<cdiv(BNb * Tg, TB), TB, 0, stream>>>(temp, flags);
  hipMemsetAsync(maxlen, 0, 4, stream);
  lens_k<<<cdiv(BNb, TB), TB, 0, stream>>>(flags, lens, maxlen);

  // ---- LSTM forward + backward ----
  for (int dir = 0; dir < 2; ++dir) {
    const float* xcur = temp;
    if (dir == 1) {
      rev_within_k<<<cdiv((long)BNb * Tg * Dd, TB), TB, 0, stream>>>(temp, lens, xrev);
      xcur = xrev;
    }
    f32_to_bf16_k<<<cdiv((long)BNb * Tg * Dd, TB), TB, 0, stream>>>(xcur, xbf, BNb * Tg * Dd);
    hipMemsetAsync(cbuf, 0, (size_t)BNb * Dd * 4, stream);
    hipMemsetAsync(hbf, 0, (size_t)BNb * Dd * 2, stream);
    float* hs = (dir == 0) ? hsF : hsB;
    const __bf16* wi = wihB + (size_t)dir * G4 * Dd;
    const __bf16* wh = whhB + (size_t)dir * G4 * Dd;
    const float* bs = bsum + (size_t)dir * G4;
    for (int t = 0; t < Tg; ++t) {
      // gates = x_t @ Wih^T + (bih+bhh)
      gemm_bf16_wmma<<<gemmL, TB, GEMM_LDS, stream>>>(xbf + (size_t)t * Dd, wi, nullptr, bs, gbuf,
                                                      BNb, G4, Dd, Tg * Dd, Dd, 0, G4);
      // gates += h @ Whh^T   (Cin aliases Out: safe, read-then-write per element)
      gemm_bf16_wmma<<<gemmL, TB, GEMM_LDS, stream>>>(hbf, wh, gbuf, nullptr, gbuf,
                                                      BNb, G4, Dd, Dd, Dd, G4, G4);
      lstm_gate_k<<<cdiv((long)BNb * Dd, TB), TB, 0, stream>>>(gbuf, cbuf, hs, hbf, t);
    }
  }
  pooled_k<<<cdiv((long)BNb * Dd, TB), TB, 0, stream>>>(hsF, hsB, lens, maxlen, pooled);

  // ---- scatter into full embedding ----
  hipMemsetAsync(bufA, 0, (size_t)Nn * Dd * 4, stream);
  scatter_k<<<cdiv((long)BNb * Dd, TB), TB, 0, stream>>>(pooled, clf_nb, bufA);

  // ---- final attention conv on edges[-1] ----
  const int* srcp = edges + (size_t)4 * 2 * Ee;
  const int* dstp = srcp + Ee;
  f32_to_bf16_k<<<cdiv((long)Nn * Dd, TB), TB, 0, stream>>>(bufA, xnbf, Nn * Dd);
  gemm_bf16_wmma<<<gemmN, TB, GEMM_LDS, stream>>>(xnbf, wbfC, nullptr, clf_b, hfeat,
                                                  Nn, Dd, Dd, Dd, Dd, 0, Dd);
  node_proj_k<<<cdiv((long)Nn * Hh, TB), TB, 0, stream>>>(hfeat, clf_asrc, clf_adst, ps, pd, Nn);
  fill_f32<<<cdiv((long)Nn * Hh, TB), TB, 0, stream>>>(mM, -1e30f, Nn * Hh);
  hipMemsetAsync(sS, 0, (size_t)Nn * Hh * 4, stream);
  edge_max_k<<<cdiv((long)Ee * Hh, TB), TB, 0, stream>>>(srcp, dstp, ps, pd, alpha, mM);
  edge_exp_k<<<cdiv((long)Ee * Hh, TB), TB, 0, stream>>>(dstp, alpha, mM, sS);
  alpha_norm_k<<<cdiv((long)Ee * Hh, TB), TB, 0, stream>>>(dstp, alpha, sS);
  hipMemsetAsync(bufB, 0, (size_t)Nn * Dd * 4, stream);
  aggregate_k<<<cdiv((long)Ee * 32, TB), TB, 0, stream>>>(srcp, dstp, alpha, hfeat, bufB);

  // ---- argmax-attention source selection ----
  fill_f32<<<cdiv(Nn, TB), TB, 0, stream>>>(maxv, -1e30f, Nn);
  fill_i32<<<cdiv(Nn, TB), TB, 0, stream>>>(argm, Ee, Nn);
  atts_k<<<cdiv(Ee, TB), TB, 0, stream>>>(dstp, alpha, atts, maxv);
  argmin_k<<<cdiv(Ee, TB), TB, 0, stream>>>(dstp, atts, maxv, argm);

  // ---- classifier head ----
  gather_rows_k<<<cdiv((long)BCc * Dd, TB), TB, 0, stream>>>(bufB, clf_nd, hclf);
  colstats_k<<<Dd, TB, 0, stream>>>(hclf, BCc, meanb, rstdb);
  mlp_k<<<BCc, TB, 0, stream>>>(hclf, meanb, rstdb, bn2_g, bn2_b, W1, b1, W2, b2, W3, b3, outp);
  srcsel_k<<<cdiv(BCc, TB), TB, 0, stream>>>(argm, clf_nd, srcp, outp + BCc);
}
